// Codebook_55954833932968
// MI455X (gfx1250) — matrix operand order repaired and checked by reading, not compiled, not run on hardware
//
#include <hip/hip_runtime.h>
#include <stdint.h>

#define K_CB   8192
#define DIM    256
#define NB     16
#define HW_SZ  1024                  // 32*32
#define NROWS  (NB * HW_SZ)          // 16384
#define BETA_F 0.25f

#define NSLICE   8                   // K split across blocks
#define MBLOCKS  32                  // 512 rows per block
#define KTILES   ((K_CB / NSLICE) / 16)   // 64 tiles per block
#define LDS_ROW  (DIM * 2 + 16)      // 528 B: pad so bank index = r*16 mod 256

typedef __bf16          v16bf  __attribute__((ext_vector_type(16)));
typedef float           v8f    __attribute__((ext_vector_type(8)));
typedef unsigned short  ush8   __attribute__((ext_vector_type(8)));
typedef unsigned short  ush16  __attribute__((ext_vector_type(16)));

union BfFrag { ush16 us; v16bf bf; };

__device__ __forceinline__ unsigned short f2bf_rne(float f) {
    union { float f; unsigned int u; } x; x.f = f;
    unsigned int r = x.u + 0x7FFFu + ((x.u >> 16) & 1u);
    return (unsigned short)(r >> 16);
}

__device__ __forceinline__ v16bf load_frag(const unsigned short* p, int second_off) {
    ush8 lo = *(const ush8*)(p);
    ush8 h8 = *(const ush8*)(p + second_off);
    BfFrag u;
    u.us = __builtin_shufflevector(lo, h8, 0,1,2,3,4,5,6,7,8,9,10,11,12,13,14,15);
    return u.bf;
}

// ---------------------------------------------------------------------------
// Kernel 1: codebook fp32 -> bf16, plus per-row squared norms.
// ---------------------------------------------------------------------------
__global__ __launch_bounds__(256)
void cb_prep(const float* __restrict__ cb,
             unsigned short* __restrict__ cb_bf,
             float* __restrict__ cbn) {
    const int k = blockIdx.x;
    const int t = threadIdx.x;
    const float f = cb[(size_t)k * DIM + t];
    cb_bf[(size_t)k * DIM + t] = f2bf_rne(f);
    float s = f * f;
    #pragma unroll
    for (int off = 16; off > 0; off >>= 1) s += __shfl_down(s, off, 32);
    __shared__ float wsum[8];
    if ((t & 31) == 0) wsum[t >> 5] = s;
    __syncthreads();
    if (t == 0) {
        float tot = 0.f;
        #pragma unroll
        for (int i = 0; i < 8; ++i) tot += wsum[i];
        cbn[k] = tot;
    }
}

// ---------------------------------------------------------------------------
// Kernel 2: z [B,C,H,W] fp32 -> z_bf [N, C] bf16 (LDS-tiled transpose).
// ---------------------------------------------------------------------------
__global__ __launch_bounds__(256)
void z_prep(const float* __restrict__ z, unsigned short* __restrict__ zbf) {
    const int tid = threadIdx.x;
    const int b   = blockIdx.x >> 8;
    const int rr  = blockIdx.x & 255;
    const int c0  = (rr >> 5) << 5;
    const int hw0 = (rr & 31) << 5;
    __shared__ float tile[32][33];
    #pragma unroll
    for (int i = 0; i < 4; ++i) {
        int t  = tid + i * 256;
        int cc = t >> 5, ww = t & 31;
        tile[cc][ww] = z[((size_t)(b * DIM + c0 + cc)) * HW_SZ + hw0 + ww];
    }
    __syncthreads();
    #pragma unroll
    for (int i = 0; i < 4; ++i) {
        int t  = tid + i * 256;
        int ww = t >> 5, cc = t & 31;
        zbf[((size_t)(b * HW_SZ + hw0 + ww)) * DIM + c0 + cc] = f2bf_rne(tile[cc][ww]);
    }
}

// ---------------------------------------------------------------------------
// Kernel 3: WMMA argmin, LDS-staged shared-B.
// Grid = 32 M-blocks x 8 K-slices. Block = 8 waves; wave w exclusively owns
// rows m0 + 64w .. +63 (4 A-tiles in registers). All waves walk the same
// K-slice; each 16-entry x 256-dim codebook tile (8 KB bf16) is staged ONCE
// per block into double-buffered LDS -> L2 codebook traffic drops 8x vs
// per-wave streaming. Score s[n,k] = ||e_k||^2 - 2*(z_n . e_k).
// ---------------------------------------------------------------------------
__global__ __launch_bounds__(256, 1)
void vq_argmin(const unsigned short* __restrict__ zbf,
               const unsigned short* __restrict__ cbbf,
               const float* __restrict__ cbn,
               float* __restrict__ pval,
               int* __restrict__ pidx) {
    const int tid   = threadIdx.x;
    const int wave  = tid >> 5;
    const int lane  = tid & 31;
    const int r     = lane & 15;     // A row-in-tile / B col-in-tile / C col
    const int hi    = lane >> 4;     // lane half selector
    const int mblk  = blockIdx.x >> 3;
    const int slice = blockIdx.x & (NSLICE - 1);
    const int m0    = mblk * 512 + wave * 64;
    const int ks0   = slice * (K_CB / NSLICE);

    __shared__ unsigned char bbuf[2][16 * LDS_ROW];   // 2 x 8448 B

    // ---- A fragments: 4 M-tiles of 16 rows (loop invariant) ----------------
    // 16-bit A 16x32 layout: per lane, 16B chunk at K-offset hi*8 and +16.
    v16bf a[4][8];
    #pragma unroll
    for (int mt = 0; mt < 4; ++mt) {
        const unsigned short* zrow = zbf + (size_t)(m0 + mt * 16 + r) * DIM;
        #pragma unroll
        for (int dc = 0; dc < 8; ++dc)
            a[mt][dc] = load_frag(zrow + dc * 32 + hi * 8, 16);
    }

    float minv[4][8];
    int   mini[4][8];
    #pragma unroll
    for (int mt = 0; mt < 4; ++mt)
        #pragma unroll
        for (int j = 0; j < 8; ++j) { minv[mt][j] = 3.402823466e38f; mini[mt][j] = 0; }

    // ---- Cooperative staging: 256 threads move one 8 KB tile ---------------
    const int srow = tid >> 4;       // codebook entry within tile
    const int schk = tid & 15;       // 32B chunk within the 512B row
    auto stage = [&](int kt, int p) {
        const unsigned short* g =
            cbbf + (size_t)(ks0 + kt * 16 + srow) * DIM + schk * 16;
        ush8 q0 = *(const ush8*)(g);
        ush8 q1 = *(const ush8*)(g + 8);
        unsigned char* d = &bbuf[p][srow * LDS_ROW + schk * 32];
        *(ush8*)(d)      = q0;
        *(ush8*)(d + 16) = q1;
    };

    stage(0, 0);
    __syncthreads();

    for (int kt = 0; kt < KTILES; ++kt) {
        const int p = kt & 1;
        if (kt + 1 < KTILES) {
            stage(kt + 1, p ^ 1);
            if (kt + 2 < KTILES)   // warm L0 for the tile after next
                __builtin_prefetch(
                    cbbf + (size_t)(ks0 + (kt + 2) * 16 + srow) * DIM + schk * 16, 0, 3);
        }

        const int k0 = ks0 + kt * 16;
        // 16-bit B 32x16 layout: per lane, 16B chunk at K-offset hi*8 and +16.
        const unsigned char* brow = &bbuf[p][r * LDS_ROW + hi * 16];

        v8f acc[4];
        #pragma unroll
        for (int mt = 0; mt < 4; ++mt)
            acc[mt] = (v8f){0.f, 0.f, 0.f, 0.f, 0.f, 0.f, 0.f, 0.f};

        #pragma unroll
        for (int dc = 0; dc < 8; ++dc) {
            v16bf bfrag = load_frag((const unsigned short*)(brow + dc * 64), 16);
            #pragma unroll
            for (int mt = 0; mt < 4; ++mt)
                acc[mt] = __builtin_amdgcn_wmma_f32_16x16x32_bf16(
                    false, a[mt][dc], false, bfrag, (short)0, acc[mt], false, false);
        }

        const float cn   = cbn[k0 + r];
        const int   kcol = k0 + r;
        #pragma unroll
        for (int mt = 0; mt < 4; ++mt)
            #pragma unroll
            for (int j = 0; j < 8; ++j) {
                float s = __builtin_fmaf(-2.0f, acc[mt][j], cn);
                if (s < minv[mt][j]) { minv[mt][j] = s; mini[mt][j] = kcol; }
            }

        __syncthreads();   // staging of p^1 done; compute on p done
    }

    // ---- Per-row argmin: shuffle reduce within each 16-lane half -----------
    // C layout: VGPR j -> row (mt*16 + j + hi*8), col r. Rows are wave-private.
    #pragma unroll
    for (int mt = 0; mt < 4; ++mt)
        #pragma unroll
        for (int j = 0; j < 8; ++j) {
            float v = minv[mt][j];
            int   i = mini[mt][j];
            #pragma unroll
            for (int off = 8; off >= 1; off >>= 1) {
                float ov = __shfl_xor(v, off, 32);
                int   oi = __shfl_xor(i, off, 32);
                if (ov < v) { v = ov; i = oi; }
            }
            if (r == 0) {
                int row = m0 + mt * 16 + j + hi * 8;
                pval[slice * NROWS + row] = v;
                pidx[slice * NROWS + row] = i;
            }
        }
}

// ---------------------------------------------------------------------------
// Kernel 3b: combine per-slice argmin candidates (8 per row).
// ---------------------------------------------------------------------------
__global__ __launch_bounds__(256)
void argmin_combine(const float* __restrict__ pval,
                    const int* __restrict__ pidx,
                    int* __restrict__ out_idx) {
    const int n = blockIdx.x * 256 + threadIdx.x;
    float bv = 3.402823466e38f;
    int   bi = 0;
    #pragma unroll
    for (int s = 0; s < NSLICE; ++s) {
        float v = pval[s * NROWS + n];
        int   i = pidx[s * NROWS + n];
        if (v < bv) { bv = v; bi = i; }
    }
    out_idx[n] = bi;
}

// ---------------------------------------------------------------------------
// Kernel 4: gather z_q (back to [B,C,H,W]) + per-block partial loss sums.
// ---------------------------------------------------------------------------
__global__ __launch_bounds__(256)
void gather_loss(const float* __restrict__ z,
                 const float* __restrict__ cb,
                 const int* __restrict__ idx,
                 float* __restrict__ zq,
                 float* __restrict__ partial) {
    const int tid = threadIdx.x;
    const int b   = blockIdx.x >> 8;
    const int rr  = blockIdx.x & 255;
    const int c0  = (rr >> 5) << 5;
    const int hw0 = (rr & 31) << 5;
    __shared__ float tile[32][33];
    __shared__ int   kk[32];
    if (tid < 32) kk[tid] = idx[b * HW_SZ + hw0 + tid];
    __syncthreads();
    #pragma unroll
    for (int i = 0; i < 4; ++i) {
        int t  = tid + i * 256;
        int ww = t >> 5, cc = t & 31;
        tile[ww][cc] = cb[(size_t)kk[ww] * DIM + c0 + cc];
    }
    __syncthreads();
    float acc = 0.f;
    #pragma unroll
    for (int i = 0; i < 4; ++i) {
        int t  = tid + i * 256;
        int cc = t >> 5, ww = t & 31;
        size_t off = ((size_t)(b * DIM + c0 + cc)) * HW_SZ + hw0 + ww;
        float e = tile[ww][cc];
        float d = e - z[off];
        zq[off] = e;
        acc += d * d;
    }
    #pragma unroll
    for (int off = 16; off > 0; off >>= 1) acc += __shfl_down(acc, off, 32);
    __shared__ float wsum[8];
    if ((tid & 31) == 0) wsum[tid >> 5] = acc;
    __syncthreads();
    if (tid == 0) {
        float tot = 0.f;
        #pragma unroll
        for (int i = 0; i < 8; ++i) tot += wsum[i];
        partial[blockIdx.x] = tot;
    }
}

// ---------------------------------------------------------------------------
// Kernel 5: final loss reduction. loss = (1+beta) * mean((z_q - z)^2).
// ---------------------------------------------------------------------------
__global__ __launch_bounds__(256)
void loss_final(const float* __restrict__ partial, float* __restrict__ out_loss) {
    const int tid = threadIdx.x;
    float s = 0.f;
    for (int i = tid; i < 4096; i += 256) s += partial[i];
    #pragma unroll
    for (int off = 16; off > 0; off >>= 1) s += __shfl_down(s, off, 32);
    __shared__ float wsum[8];
    if ((tid & 31) == 0) wsum[tid >> 5] = s;
    __syncthreads();
    if (tid == 0) {
        float tot = 0.f;
        #pragma unroll
        for (int i = 0; i < 8; ++i) tot += wsum[i];
        *out_loss = (1.0f + BETA_F) * tot / (float)((size_t)NROWS * DIM);
    }
}

// ---------------------------------------------------------------------------
extern "C" void kernel_launch(void* const* d_in, const int* in_sizes, int n_in,
                              void* d_out, int out_size, void* d_ws, size_t ws_size,
                              hipStream_t stream) {
    const float* z  = (const float*)d_in[0];   // [16,256,32,32]
    const float* cb = (const float*)d_in[1];   // [8192,256]

    float* out     = (float*)d_out;
    float* zq      = out;                                  // 4194304 f32
    float* loss    = out + (size_t)NROWS * DIM;            // 1 f32
    int*   out_idx = (int*)(out + (size_t)NROWS * DIM + 1);// 16384 i32

    char* w = (char*)d_ws;
    size_t off = 0;
    unsigned short* cb_bf = (unsigned short*)(w + off); off += (size_t)K_CB * DIM * 2;   // 4 MB
    float*          cbn   = (float*)(w + off);          off += (size_t)K_CB * 4;         // 32 KB
    unsigned short* z_bf  = (unsigned short*)(w + off); off += (size_t)NROWS * DIM * 2;  // 8 MB
    float*          partial = (float*)(w + off);        off += (size_t)NB * 256 * 4;     // 16 KB
    float*          pval  = (float*)(w + off);          off += (size_t)NSLICE * NROWS * 4; // 512 KB
    int*            pidx  = (int*)(w + off);            off += (size_t)NSLICE * NROWS * 4; // 512 KB

    cb_prep       <<<K_CB,             256, 0, stream>>>(cb, cb_bf, cbn);
    z_prep        <<<NB * 256,         256, 0, stream>>>(z, z_bf);
    vq_argmin     <<<MBLOCKS * NSLICE, 256, 0, stream>>>(z_bf, cb_bf, cbn, pval, pidx);
    argmin_combine<<<NROWS / 256,      256, 0, stream>>>(pval, pidx, out_idx);
    gather_loss   <<<NB * 256,         256, 0, stream>>>(z, cb, out_idx, zq, partial);
    loss_final    <<<1,                256, 0, stream>>>(partial, loss);
}
